// TH_LSTM_65592740545325
// MI455X (gfx1250) — matrix operand order repaired and checked by reading, not compiled, not run on hardware
//
#include <hip/hip_runtime.h>
#include <hip/hip_bf16.h>

// ---------------------------------------------------------------------------
// TH-LSTM for MI455X (gfx1250), wave32 + WMMA f32_16x16x32_f16 +
// GLOBAL_LOAD_ASYNC_TO_LDS_B128 double-buffered staging (ASYNCcnt pipeline).
// B=512, T=128, D=128, H=512, MEM=64.
//   * Per step: GEMM1 [512x640]x[640x3072] (A = [h_prev | x_t] split-K) ->
//     s_t activation -> GEMM2 [512x512]x[512x2560] -> gate combine.
//   * Attention: q@Waq, hs@Wah (M=32768) WMMA GEMMs -> wave-reduced scores ->
//     softmax+weighted sum -> three accumulated WMMA GEMMs -> sigmoid head.
// f16 operands, fp32 WMMA accumulation. Deterministic, no atomics, all
// launches on `stream`, scratch only in d_ws (~138 MiB).
// ---------------------------------------------------------------------------

typedef _Float16 f16;
typedef __attribute__((ext_vector_type(16))) _Float16 v16h;
typedef __attribute__((ext_vector_type(8)))  _Float16 v8h;
typedef __attribute__((ext_vector_type(8)))  float    v8f;

union F16x16 { v16h v; v8h h[2]; };

#define TB   512   // batch
#define TT   128   // seq len
#define TD   128   // feature dim
#define TH   512   // hidden
#define TMEM 64
#define N1   3072  // 6*H
#define N2   2560  // 5*H
#define K1   640   // H + D

#if defined(__gfx1250__)
#define USE_ASYNC 1
#else
#define USE_ASYNC 0
#endif

#if USE_ASYNC
typedef __attribute__((address_space(3))) const f16 f16_lds;
// LDS byte offset of a __shared__ pointer (generic->local addrspacecast).
__device__ __forceinline__ unsigned lds_off(const f16* p) {
  return (unsigned)(unsigned long long)(f16_lds*)p;
}
// 16B global -> LDS async copy, tracked by ASYNCcnt (no VGPR round-trip).
__device__ __forceinline__ void async_cp_b128(const f16* gp, const f16* lp) {
  asm volatile("global_load_async_to_lds_b128 %0, %1, off"
               :: "v"(lds_off(lp)), "v"(gp) : "memory");
}
__device__ __forceinline__ void wait_async0() {
  asm volatile("s_wait_asynccnt 0x0" ::: "memory");
}
__device__ __forceinline__ void wait_async4() {
  asm volatile("s_wait_asynccnt 0x4" ::: "memory");
}
#endif

// ---------------- WMMA fragment loaders (ISA 7.12.2 layouts) ----------------
// A 16x32 f16: lanes 0-15 hold K {0..7,16..23}; lanes 16-31 hold K {8..15,24..31}
__device__ __forceinline__ v16h load_fragA(const f16* row, int lane) {
  int hi = (lane >> 4) * 8;
  F16x16 r;
  r.h[0] = *(const v8h*)(row + hi);
  r.h[1] = *(const v8h*)(row + 16 + hi);
  return r.v;
}
// B 32x16 f16 (from Bt row = N): lanes 0-15 hold K {0..7,16..23}; lanes 16-31 K {8..15,24..31}
__device__ __forceinline__ v16h load_fragB(const f16* row, int lane) {
  int hi = (lane >> 4) * 8;
  F16x16 r;
  r.h[0] = *(const v8h*)(row + hi);
  r.h[1] = *(const v8h*)(row + 16 + hi);
  return r.v;
}

// ---------------- Generic f16 WMMA GEMM: C(MxN,f32) = A(MxK) * Bt(NxK)^T ----
// A split at kSplit between two sources (for [h|x] and [h_last|c_last]).
// M, N multiples of 128; K, kSplit multiples of 32. 256 threads = 8 waves.
#define KC    32
#define SPAD  40   // LDS row stride in halves (padded vs 32 to spread banks)

__global__ __launch_bounds__(256) void gemm_f16_wmma(
    const f16* __restrict__ A0, int lda0,
    const f16* __restrict__ A1, int lda1, int kSplit,
    const f16* __restrict__ Bt,          // N x K row-major
    const float* __restrict__ bias,      // length N or null
    float* __restrict__ C,               // M x N row-major
    int M, int N, int K, int accumulate)
{
  __shared__ f16 sA[2][128 * SPAD];
  __shared__ f16 sB[2][128 * SPAD];

  const int tid   = threadIdx.x;
  const int lane  = tid & 31;
  const int wid   = tid >> 5;     // 0..7
  const int waveM = wid & 3;      // 4 waves along M (32 rows each)
  const int waveN = wid >> 2;     // 2 waves along N (64 cols each)
  const int bM    = blockIdx.x * 128;
  const int bN    = blockIdx.y * 128;
  const int l15   = lane & 15;

  v8f acc[2][4] = {};

  // Stage one 128x32 A tile + 128x32 Bt tile into LDS buffer `buf`.
  // Each thread handles 2x 16B for A and 2x 16B for B (4 async ops/wave/chunk).
  auto stage = [&](int k0, int buf) {
    const f16* aSrc; int alda;
    if (k0 < kSplit) { aSrc = A0 + k0;            alda = lda0; }
    else             { aSrc = A1 + (k0 - kSplit); alda = lda1; }
    for (int it = 0; it < 2; ++it) {
      int e  = it * 256 + tid;
      int r  = e >> 2;
      int cv = (e & 3) * 8;
      const f16* ga = aSrc + (long)(bM + r) * alda + cv;
      const f16* gb = Bt + (long)(bN + r) * K + k0 + cv;
#if USE_ASYNC
      async_cp_b128(ga, &sA[buf][r * SPAD + cv]);
      async_cp_b128(gb, &sB[buf][r * SPAD + cv]);
#else
      *(v8h*)(&sA[buf][r * SPAD + cv]) = *(const v8h*)ga;
      *(v8h*)(&sB[buf][r * SPAD + cv]) = *(const v8h*)gb;
#endif
    }
  };

  const int nk = K / KC;
  stage(0, 0);
  for (int i = 0; i < nk; ++i) {
    int buf = i & 1;
    if (i + 1 < nk) stage((i + 1) * KC, buf ^ 1);
#if USE_ASYNC
    // Each chunk = 4 async ops per wave, completing in order: waiting <=4
    // guarantees chunk i landed while chunk i+1 stays in flight.
    if (i + 1 < nk) wait_async4(); else wait_async0();
#endif
    __syncthreads();

    v16h af[2], bf[4];
    for (int mi = 0; mi < 2; ++mi)
      af[mi] = load_fragA(&sA[buf][(waveM * 32 + mi * 16 + l15) * SPAD], lane);
    for (int ni = 0; ni < 4; ++ni)
      bf[ni] = load_fragB(&sB[buf][(waveN * 64 + ni * 16 + l15) * SPAD], lane);

    for (int mi = 0; mi < 2; ++mi)
      for (int ni = 0; ni < 4; ++ni)
        acc[mi][ni] = __builtin_amdgcn_wmma_f32_16x16x32_f16(
            false, af[mi], false, bf[ni], (short)0, acc[mi][ni], false, false);
    __syncthreads();   // all waves done reading `buf` before it is re-staged
  }

  // Epilogue: D element (vgpr i, lane) -> row base+(lane>=16?8:0)+i, col lane&15
  const int hiRow = (lane >> 4) * 8;
  for (int mi = 0; mi < 2; ++mi) {
    int rowBase = bM + waveM * 32 + mi * 16 + hiRow;
    for (int ni = 0; ni < 4; ++ni) {
      int col = bN + waveN * 64 + ni * 16 + l15;
      float bv = bias ? bias[col] : 0.0f;
      for (int i = 0; i < 8; ++i) {
        long idx = (long)(rowBase + i) * N + col;
        float v = acc[mi][ni][i] + bv;
        if (accumulate) v += C[idx];
        C[idx] = v;
      }
    }
  }
}

// ------------------------------ prep kernels -------------------------------
__global__ void prep_w1(const float* __restrict__ Wh, const float* __restrict__ Wx,
                        f16* __restrict__ W1t) {   // [N1][K1]
  int idx = blockIdx.x * blockDim.x + threadIdx.x;
  if (idx >= N1 * K1) return;
  int n = idx / K1, k = idx % K1;
  int g = n >> 9, h = n & 511;
  float v = (k < TH) ? Wh[((long)g * TH + k) * TH + h]
                     : Wx[((long)g * TD + (k - TH)) * TH + h];
  W1t[idx] = (f16)v;
}

__global__ void prep_w2(const float* __restrict__ Ws, f16* __restrict__ W2t) { // [N2][TH]
  int idx = blockIdx.x * blockDim.x + threadIdx.x;
  if (idx >= N2 * TH) return;
  int n = idx / TH, k = idx % TH;
  int g = n >> 9, h = n & 511;
  W2t[idx] = (f16)Ws[((long)g * TH + k) * TH + h];
}

__global__ void transpose_f16(const float* __restrict__ W, f16* __restrict__ Wt,
                              int K, int N) {      // W: KxN -> Wt: NxK
  int idx = blockIdx.x * blockDim.x + threadIdx.x;
  if (idx >= K * N) return;
  int n = idx / K, k = idx % K;
  Wt[idx] = (f16)W[(long)k * N + n];
}

__global__ void add3(const float* a, const float* b, const float* c,
                     float* o, int n) {
  int i = blockIdx.x * blockDim.x + threadIdx.x;
  if (i < n) o[i] = a[i] + b[i] + c[i];
}
__global__ void add4(const float* a, const float* b, const float* c,
                     const float* d, float* o, int n) {
  int i = blockIdx.x * blockDim.x + threadIdx.x;
  if (i < n) o[i] = a[i] + b[i] + c[i] + d[i];
}

__global__ void prep_xh(const float* __restrict__ X, f16* __restrict__ Xh) {
  int idx = blockIdx.x * blockDim.x + threadIdx.x;       // over B*T*D
  if (idx >= TB * TT * TD) return;
  int b = idx / (TT * TD), rem = idx % (TT * TD);
  int t = rem / TD, d = rem % TD;
  Xh[idx] = (f16)X[((long)b * TT + t) * (TD + 1) + d];
}
__global__ void prep_dt(const float* __restrict__ X, float* __restrict__ dv) {
  int idx = blockIdx.x * blockDim.x + threadIdx.x;       // over B*T
  if (idx >= TB * TT) return;
  int b = idx / TT, t = idx % TT;
  dv[idx] = X[((long)b * TT + t) * (TD + 1) + TD];
}

__global__ void fill_zero_f16(f16* p, int n) {
  int i = blockIdx.x * blockDim.x + threadIdx.x;
  if (i < n) p[i] = (f16)0.0f;
}
__global__ void fill_zero_f32(float* p, int n) {
  int i = blockIdx.x * blockDim.x + threadIdx.x;
  if (i < n) p[i] = 0.0f;
}

// ------------------------------ step kernels -------------------------------
__device__ __forceinline__ float sigf(float x) { return 1.0f / (1.0f + __expf(-x)); }

// s_t = tanh(pre[:, 0:512] + d_t*Wst + bst)   (bh+bx+bg[0] already in pre bias)
__global__ void act_s(const float* __restrict__ pre, const float* __restrict__ dv,
                      const float* __restrict__ Wst, const float* __restrict__ bst,
                      f16* __restrict__ sf, int t) {
  int idx = blockIdx.x * blockDim.x + threadIdx.x;       // over B*H
  if (idx >= TB * TH) return;
  int b = idx >> 9, h = idx & 511;
  float v = pre[(long)b * N1 + h] + dv[b * TT + t] * Wst[h] + bst[h];
  sf[idx] = (f16)tanhf(v);
}

__global__ void gates(const float* __restrict__ pre, const float* __restrict__ pres,
                      const f16* __restrict__ sf, float* __restrict__ c,
                      f16* __restrict__ hf, f16* __restrict__ hist,
                      f16* __restrict__ hl16, f16* __restrict__ cl16, int t) {
  int idx = blockIdx.x * blockDim.x + threadIdx.x;       // over B*H
  if (idx >= TB * TH) return;
  int b = idx >> 9, h = idx & 511;
  const float* pr = pre  + (long)b * N1;
  const float* ps = pres + (long)b * N2;
  float ft = sigf (pr[ 512 + h] + ps[        h]);
  float it = sigf (pr[1024 + h] + ps[ 512  + h]);
  float Tt = sigf (pr[1536 + h] + ps[1024  + h]);
  float zt = tanhf(pr[2048 + h] + ps[1536  + h]);
  float ot = sigf (pr[2560 + h] + ps[2048  + h]);
  float st = (float)sf[idx];
  float cn = ft * c[idx] + it * zt + Tt * st;
  c[idx] = cn;
  float hn = ot * tanhf(cn);
  hf[idx] = (f16)hn;
  if (t >= TT - TMEM) hist[((long)b * TMEM + (t - (TT - TMEM))) * TH + h] = (f16)hn;
  if (t == TT - 1) { hl16[idx] = (f16)hn; cl16[idx] = (f16)cn; }
}

// ---------------------------- attention kernels ----------------------------
// score[b,m] = exp( sum_h tanh(qa[b,h] + hw[b*64+m,h]) * vt[h] ), wave per row
__global__ void att_score(const float* __restrict__ qa, const float* __restrict__ hw,
                          const float* __restrict__ vt, float* __restrict__ score) {
  int gid  = (blockIdx.x * blockDim.x + threadIdx.x) >> 5;
  int lane = threadIdx.x & 31;
  if (gid >= TB * TMEM) return;
  int b = gid >> 6;
  float acc = 0.0f;
  for (int h = lane; h < TH; h += 32)
    acc += tanhf(qa[(long)b * TH + h] + hw[(long)gid * TH + h]) * vt[h];
  for (int off = 16; off; off >>= 1) acc += __shfl_down(acc, off, 32);
  if (lane == 0) score[gid] = __expf(acc);
}

// e[b,h] = sum_m (score/sum) * hist[b,m,h]; one block per b
__global__ void att_e(const float* __restrict__ score, const f16* __restrict__ hist,
                      f16* __restrict__ e16) {
  int b = blockIdx.x, tid = threadIdx.x;
  float sum = 0.0f;
  for (int m = 0; m < TMEM; ++m) sum += score[b * TMEM + m];
  float inv = 1.0f / sum;
  for (int h = tid; h < TH; h += 256) {
    float acc = 0.0f;
    for (int m = 0; m < TMEM; ++m)
      acc += score[b * TMEM + m] * (float)hist[((long)b * TMEM + m) * TH + h];
    e16[(long)b * TH + h] = (f16)(acc * inv);
  }
}

// out[b] = sigmoid( sum_h tanh(fin[b,h]) * Wc[h] + bc ); one wave per b
__global__ void final_out(const float* __restrict__ fin, const float* __restrict__ Wc,
                          const float* __restrict__ bc, float* __restrict__ out) {
  int b = blockIdx.x, lane = threadIdx.x;
  float acc = 0.0f;
  for (int h = lane; h < TH; h += 32) acc += tanhf(fin[(long)b * TH + h]) * Wc[h];
  for (int off = 16; off; off >>= 1) acc += __shfl_down(acc, off, 32);
  if (lane == 0) out[b] = 1.0f / (1.0f + __expf(-(acc + bc[0])));
}

// ------------------------------- launcher ----------------------------------
extern "C" void kernel_launch(void* const* d_in, const int* in_sizes, int n_in,
                              void* d_out, int out_size, void* d_ws, size_t ws_size,
                              hipStream_t stream) {
  (void)in_sizes; (void)n_in; (void)out_size; (void)ws_size;
  const float* X      = (const float*)d_in[0];
  const float* Wh     = (const float*)d_in[1];
  const float* bh_lin = (const float*)d_in[2];
  const float* Wx     = (const float*)d_in[3];
  const float* bx_lin = (const float*)d_in[4];
  const float* Wst    = (const float*)d_in[5];
  const float* bst    = (const float*)d_in[6];
  const float* Ws     = (const float*)d_in[7];
  const float* bs_lin = (const float*)d_in[8];
  const float* bg     = (const float*)d_in[9];
  const float* Waq    = (const float*)d_in[10];
  const float* baq    = (const float*)d_in[11];
  const float* Wah    = (const float*)d_in[12];
  const float* bah    = (const float*)d_in[13];
  const float* vt     = (const float*)d_in[14];
  const float* W_h    = (const float*)d_in[15];
  const float* b_Wh   = (const float*)d_in[16];
  const float* We     = (const float*)d_in[17];
  const float* b_We   = (const float*)d_in[18];
  const float* Wg     = (const float*)d_in[19];
  const float* b_Wg   = (const float*)d_in[20];
  const float* bh_p   = (const float*)d_in[21];
  const float* Wc     = (const float*)d_in[22];
  const float* bc     = (const float*)d_in[23];
  float* out = (float*)d_out;

  // ---- carve workspace (~138 MiB) ----
  char* p = (char*)d_ws;
  auto carve = [&](size_t bytes) -> void* {
    void* r = (void*)p;
    p += (bytes + 255) & ~(size_t)255;
    return r;
  };
  f16*   W1t   = (f16*)  carve((size_t)N1 * K1 * 2);
  f16*   W2t   = (f16*)  carve((size_t)N2 * TH * 2);
  f16*   Waqt  = (f16*)  carve((size_t)TH * 2 * TH * 2);   // 512 x 1024
  f16*   Waht  = (f16*)  carve((size_t)TH * TH * 2);
  f16*   W_ht  = (f16*)  carve((size_t)TH * TH * 2);
  f16*   Wet   = (f16*)  carve((size_t)TH * TH * 2);
  f16*   Wgt   = (f16*)  carve((size_t)TH * TD * 2);
  float* bias1 = (float*)carve((size_t)N1 * 4);
  float* biasF = (float*)carve((size_t)TH * 4);
  f16*   Xh    = (f16*)  carve((size_t)TB * TT * TD * 2);
  float* dv    = (float*)carve((size_t)TB * TT * 4);
  f16*   hf    = (f16*)  carve((size_t)TB * TH * 2);
  f16*   sf    = (f16*)  carve((size_t)TB * TH * 2);
  float* cbuf  = (float*)carve((size_t)TB * TH * 4);
  float* pre   = (float*)carve((size_t)TB * N1 * 4);
  float* pres  = (float*)carve((size_t)TB * N2 * 4);
  f16*   hist  = (f16*)  carve((size_t)TB * TMEM * TH * 2);
  f16*   hl16  = (f16*)  carve((size_t)TB * TH * 2);
  f16*   cl16  = (f16*)  carve((size_t)TB * TH * 2);
  float* qa    = (float*)carve((size_t)TB * TH * 4);
  float* hw    = (float*)carve((size_t)TB * TMEM * TH * 4);  // 64 MB
  float* score = (float*)carve((size_t)TB * TMEM * 4);
  f16*   e16   = (f16*)  carve((size_t)TB * TH * 2);
  float* fin   = (float*)carve((size_t)TB * TH * 4);

  auto blocks = [](long n) { return (unsigned)((n + 255) / 256); };
  auto gemm = [&](const f16* A0, int lda0, const f16* A1, int lda1, int kSplit,
                  const f16* Bt, const float* bias, float* C,
                  int M, int N, int K, int accum) {
    gemm_f16_wmma<<<dim3(M / 128, N / 128), 256, 0, stream>>>(
        A0, lda0, A1, lda1, kSplit, Bt, bias, C, M, N, K, accum);
  };

  // ---- prep: f16 transposed weights, fused biases, x/dt split, zero state ----
  prep_w1<<<blocks((long)N1 * K1), 256, 0, stream>>>(Wh, Wx, W1t);
  prep_w2<<<blocks((long)N2 * TH), 256, 0, stream>>>(Ws, W2t);
  transpose_f16<<<blocks((long)2 * TH * TH), 256, 0, stream>>>(Waq, Waqt, 2 * TH, TH);
  transpose_f16<<<blocks((long)TH * TH), 256, 0, stream>>>(Wah, Waht, TH, TH);
  transpose_f16<<<blocks((long)TH * TH), 256, 0, stream>>>(W_h, W_ht, TH, TH);
  transpose_f16<<<blocks((long)TH * TH), 256, 0, stream>>>(We,  Wet,  TH, TH);
  transpose_f16<<<blocks((long)TD * TH), 256, 0, stream>>>(Wg,  Wgt,  TD, TH);
  add3<<<blocks(N1), 256, 0, stream>>>(bh_lin, bx_lin, bg, bias1, N1);
  add4<<<blocks(TH), 256, 0, stream>>>(b_Wh, b_We, b_Wg, bh_p, biasF, TH);
  prep_xh<<<blocks((long)TB * TT * TD), 256, 0, stream>>>(X, Xh);
  prep_dt<<<blocks((long)TB * TT), 256, 0, stream>>>(X, dv);
  fill_zero_f16<<<blocks((long)TB * TH), 256, 0, stream>>>(hf, TB * TH);
  fill_zero_f32<<<blocks((long)TB * TH), 256, 0, stream>>>(cbuf, TB * TH);

  // ---- recurrence: 128 steps x (GEMM1, s, GEMM2, gates) ----
  for (int t = 0; t < TT; ++t) {
    // pre = [h_prev | x_t] @ [Wh;Wx]  (+ bh+bx+bg bias), M=512 N=3072 K=640
    gemm(hf, TH, Xh + (long)t * TD, TT * TD, TH, W1t, bias1, pre, TB, N1, K1, 0);
    act_s<<<blocks((long)TB * TH), 256, 0, stream>>>(pre, dv, Wst, bst, sf, t);
    // pres = s_t @ Ws  (+ bs_lin bias), M=512 N=2560 K=512
    gemm(sf, TH, sf, TH, TH, W2t, bs_lin, pres, TB, N2, TH, 0);
    gates<<<blocks((long)TB * TH), 256, 0, stream>>>(pre, pres, sf, cbuf,
                                                     hf, hist, hl16, cl16, t);
  }

  // ---- attention ----
  // qa = [h_last | c_last] @ Waq + baq, M=512 N=512 K=1024
  gemm(hl16, TH, cl16, TH, TH, Waqt, baq, qa, TB, TH, 2 * TH, 0);
  // hw = hs @ Wah + bah, M=32768 N=512 K=512
  gemm(hist, TH, hist, TH, TH, Waht, bah, hw, TB * TMEM, TH, TH, 0);
  att_score<<<(TB * TMEM * 32) / 256, 256, 0, stream>>>(qa, hw, vt, score);
  att_e<<<TB, 256, 0, stream>>>(score, hist, e16);

  // ---- final head: fin = h_last@W_h + e@We + x_last@Wg + (all biases) ----
  gemm(hl16, TH, hl16, TH, TH, W_ht, biasF, fin, TB, TH, TH, 0);
  gemm(e16,  TH, e16,  TH, TH, Wet,  nullptr, fin, TB, TH, TH, 1);
  gemm(Xh + (long)(TT - 1) * TD, TT * TD, Xh, TT * TD, TD, Wgt, nullptr,
       fin, TB, TH, TD, 1);
  final_out<<<TB, 32, 0, stream>>>(fin, Wc, bc, out);
}
